// multHeadMILAtten_12859132084753
// MI455X (gfx1250) — matrix operand order repaired and checked by reading, not compiled, not run on hardware
//
#include <hip/hip_runtime.h>
#include <hip/hip_bf16.h>
#include <math.h>

#define DIMK   512
#define NHEAD  8
#define DM     64
#define DL     128
#define BB     8
#define NN     4096
#define NROW   (BB * NN)            // 32768 instance rows
#define NCOLW  (NHEAD * DM + DL)    // 640 fused output columns (512 mh + 128 gate)

typedef __bf16 bf16_t;
typedef __bf16 bf16x8 __attribute__((ext_vector_type(8)));
typedef __bf16 v16bf  __attribute__((ext_vector_type(16)));
typedef float  v8f    __attribute__((ext_vector_type(8)));

union Frag16 {
    v16bf  v;
    bf16x8 h[2];
};

#if __has_builtin(__builtin_amdgcn_global_load_async_to_lds_b128) && \
    __has_builtin(__builtin_amdgcn_s_wait_asynccnt)
#define USE_ASYNC_LDS 1
#else
#define USE_ASYNC_LDS 0
#endif

#if USE_ASYNC_LDS
#define ASYNC_WAIT(n) __builtin_amdgcn_s_wait_asynccnt(n)
#else
#define ASYNC_WAIT(n)
#endif

// Exact pointee type the builtin expects: 16-byte int vector.
typedef int i4v __attribute__((vector_size(16)));
#define GPTR(p) ((__attribute__((address_space(1))) i4v*)(p))
#define LPTR(p) ((__attribute__((address_space(3))) i4v*)(p))

// Copy 64 bytes (32 bf16) global -> LDS; async DMA when available.
__device__ __forceinline__ void cp_tile64(const unsigned short* g, bf16_t* l) {
#if USE_ASYNC_LDS
    __builtin_amdgcn_global_load_async_to_lds_b128(GPTR(g), LPTR(l),  0, 0);
    __builtin_amdgcn_global_load_async_to_lds_b128(GPTR(g), LPTR(l), 16, 0);
    __builtin_amdgcn_global_load_async_to_lds_b128(GPTR(g), LPTR(l), 32, 0);
    __builtin_amdgcn_global_load_async_to_lds_b128(GPTR(g), LPTR(l), 48, 0);
#else
    const uint4* s = (const uint4*)g;
    uint4*       d = (uint4*)l;
#pragma unroll
    for (int i = 0; i < 4; ++i) d[i] = s[i];
#endif
}

__device__ __forceinline__ unsigned short f32_to_bf16_bits(float f) {
    unsigned u = __builtin_bit_cast(unsigned, f);
    u += 0x7FFFu + ((u >> 16) & 1u);   // round-to-nearest-even
    return (unsigned short)(u >> 16);
}

// ---------------------------------------------------------------------------
// K0a: x (f32) -> xbf (bf16), row-major [r][K]. 4 elements/thread.
// ---------------------------------------------------------------------------
__global__ void prep_x(const float* __restrict__ x, unsigned short* __restrict__ xbf) {
    size_t i = ((size_t)blockIdx.x * blockDim.x + threadIdx.x) * 4;
    float4 f = *(const float4*)(x + i);
    ushort4 h;
    h.x = f32_to_bf16_bits(f.x);
    h.y = f32_to_bf16_bits(f.y);
    h.z = f32_to_bf16_bits(f.z);
    h.w = f32_to_bf16_bits(f.w);
    *(ushort4*)(xbf + i) = h;
}

// ---------------------------------------------------------------------------
// K0b: weights -> bf16, transposed to [col][K] (col 0..511 = h*64+m from WX,
// col 512..639 = gate_w column).
// ---------------------------------------------------------------------------
__global__ void prep_weights(const float* __restrict__ WX,
                             const float* __restrict__ gate_w,
                             unsigned short* __restrict__ WT) {
    int idx = blockIdx.x * blockDim.x + threadIdx.x;
    if (idx >= NCOLW * DIMK) return;
    int c = idx >> 9;          // / 512
    int k = idx & 511;
    float val;
    if (c < NHEAD * DM) {
        int h = c >> 6, m = c & 63;
        val = WX[((size_t)h * DIMK + k) * DM + m];
    } else {
        val = gate_w[(size_t)k * DL + (c - NHEAD * DM)];
    }
    WT[(size_t)c * DIMK + k] = f32_to_bf16_bits(val);
}

// ---------------------------------------------------------------------------
// K1: Y[32768][640] = xbf @ WT^T via v_wmma_f32_16x16x32_bf16.
// 8 waves x 16 rows = 128 rows/WG; 5 column blocks of 128.
// Double-buffered K-tiles staged with async global->LDS DMA (ASYNCcnt).
// ---------------------------------------------------------------------------
__global__ __launch_bounds__(256)
void gemm_mh_gate(const unsigned short* __restrict__ xbf,
                  const unsigned short* __restrict__ WTu,
                  float* __restrict__ mh,
                  float* __restrict__ gate) {
    __shared__ __align__(16) bf16_t As[2][128 * 64];
    __shared__ __align__(16) bf16_t Bs[2][128 * 64];

    const int tid   = threadIdx.x;
    const int wave  = tid >> 5;
    const int lane  = tid & 31;
    const int row0  = blockIdx.x * 128;

    const int halfk = tid & 1;        // which 32-wide K half this thread stages
    const int lrow  = tid >> 1;       // 0..127 loader row/col
    const int loff  = lrow * 64 + halfk * 32;

    const int mrow  = lane & 15;
    const int akoff = (lane < 16) ? 0 : 8;    // A: chunks at +0/+16 (ISA 16-bit A layout)
    const int bkoff = (lane < 16) ? 0 : 8;    // B: chunks at +0/+16 (same as A)

    const unsigned short* xg = xbf + (size_t)(row0 + lrow) * DIMK + halfk * 32;

    for (int cb = 0; cb < 5; ++cb) {
        v8f acc[8];
#pragma unroll
        for (int nb = 0; nb < 8; ++nb)
#pragma unroll
            for (int e = 0; e < 8; ++e) acc[nb][e] = 0.f;

        const unsigned short* wg = WTu + (size_t)(cb * 128 + lrow) * DIMK + halfk * 32;

        // prologue: stage K-tile 0 into buffer 0
        cp_tile64(xg, &As[0][loff]);
        cp_tile64(wg, &Bs[0][loff]);

        for (int kt = 0; kt < 8; ++kt) {          // K tiles of 64 over DIMK=512
            const int p = kt & 1;
            if (kt < 7) {
                cp_tile64(xg + (kt + 1) * 64, &As[p ^ 1][loff]);
                cp_tile64(wg + (kt + 1) * 64, &Bs[p ^ 1][loff]);
                ASYNC_WAIT(8);   // 8 newer loads in flight -> tile kt complete
            } else {
                ASYNC_WAIT(0);
            }
            __syncthreads();

#pragma unroll
            for (int ks = 0; ks < 64; ks += 32) {
                Frag16 a;
                a.h[0] = *(const bf16x8*)&As[p][(wave * 16 + mrow) * 64 + ks + akoff];
                a.h[1] = *(const bf16x8*)&As[p][(wave * 16 + mrow) * 64 + ks + akoff + 16];
#pragma unroll
                for (int nb = 0; nb < 8; ++nb) {
                    Frag16 b;
                    b.h[0] = *(const bf16x8*)&Bs[p][(nb * 16 + mrow) * 64 + ks + bkoff];
                    b.h[1] = *(const bf16x8*)&Bs[p][(nb * 16 + mrow) * 64 + ks + bkoff + 16];
                    acc[nb] = __builtin_amdgcn_wmma_f32_16x16x32_bf16(
                        false, a.v, false, b.v, (short)0, acc[nb], false, false);
                }
            }
            __syncthreads();
        }

        // ---- epilogue: C layout VGPR j = (M=j | j+8), lane&15 = N ----
#pragma unroll
        for (int nb = 0; nb < 8; ++nb) {
            const int c = cb * 128 + nb * 16 + (lane & 15);
#pragma unroll
            for (int j = 0; j < 8; ++j) {
                const int rl = wave * 16 + ((lane < 16) ? j : j + 8);
                const int r  = row0 + rl;
                const float vv = acc[nb][j];
                if (c < NHEAD * DM) {
                    const int h = c >> 6, m = c & 63;
                    const int b = r >> 12, n = r & 4095;
                    mh[((((size_t)h * BB + b) * NN + n) << 6) + m] = vv;
                } else {
                    const int l = c - NHEAD * DM;
                    gate[(size_t)r * DL + l] = 1.f / (1.f + __expf(-vv));
                }
            }
        }
    }
}

// ---------------------------------------------------------------------------
// K2: scores[h][r] = sum_l tanh(mh_row . V[:,l]) * gate[r][l] * W[h][l], masked.
// ---------------------------------------------------------------------------
__global__ __launch_bounds__(256)
void scores_kernel(const float* __restrict__ mh, const float* __restrict__ gate,
                   const float* __restrict__ V, const float* __restrict__ W,
                   const int* __restrict__ true_num, float* __restrict__ scores) {
    __shared__ float Vs[DM * DL];
    __shared__ float Ws[DL];
    const int tid = threadIdx.x;
    const int h   = blockIdx.x >> 7;                 // 128 blocks per head
    const int r   = ((blockIdx.x & 127) << 8) + tid; // 0..32767

    for (int i = tid; i < DM * DL; i += 256) Vs[i] = V[i];
    if (tid < DL) Ws[tid] = W[h * DL + tid];
    __syncthreads();

    const float* mrow = mh + ((size_t)h * NROW + r) * DM;
    float4 mr[16];
#pragma unroll
    for (int i = 0; i < 16; ++i) mr[i] = ((const float4*)mrow)[i];

    const float* grow = gate + (size_t)r * DL;
    float s = 0.f;
    for (int l = 0; l < DL; ++l) {
        float d = 0.f;
#pragma unroll
        for (int i = 0; i < 16; ++i) {
            const float* vc = Vs + (i * 4) * DL + l;
            d += mr[i].x * vc[0] + mr[i].y * vc[DL] +
                 mr[i].z * vc[2 * DL] + mr[i].w * vc[3 * DL];
        }
        s += tanhf(d) * grow[l] * Ws[l];
    }
    const int b = r >> 12, n = r & 4095;
    if (n >= true_num[b]) s = -1e20f;
    scores[(size_t)h * NROW + r] = s;
}

// ---------------------------------------------------------------------------
// K3: per-(h,b) masked softmax over N + attention pooling; emits feat and soft[0].
// ---------------------------------------------------------------------------
__global__ __launch_bounds__(256)
void softmax_pool(const float* __restrict__ mh, const float* __restrict__ scores,
                  float* __restrict__ out) {
    __shared__ float red[256];
    const int tid = threadIdx.x;
    const int h = blockIdx.x >> 3, b = blockIdx.x & 7;
    const float* sc = scores + ((size_t)h * BB + b) * NN;

    float lm = -INFINITY;
    for (int n = tid; n < NN; n += 256) lm = fmaxf(lm, sc[n]);
    red[tid] = lm; __syncthreads();
    for (int s = 128; s > 0; s >>= 1) {
        if (tid < s) red[tid] = fmaxf(red[tid], red[tid + s]);
        __syncthreads();
    }
    const float smax = red[0]; __syncthreads();

    float ls = 0.f;
    for (int n = tid; n < NN; n += 256) ls += __expf(sc[n] - smax);
    red[tid] = ls; __syncthreads();
    for (int s = 128; s > 0; s >>= 1) {
        if (tid < s) red[tid] += red[tid + s];
        __syncthreads();
    }
    const float inv = 1.f / red[0]; __syncthreads();

    if (h == 0) {
        float* soft0 = out + BB * NHEAD * DM + (size_t)b * NN;
        for (int n = tid; n < NN; n += 256) soft0[n] = __expf(sc[n] - smax) * inv;
    }

    const int g = tid >> 6, m = tid & 63;
    const float* mhb = mh + ((size_t)h * BB + b) * NN * DM;
    float acc = 0.f;
    for (int n = g; n < NN; n += 4)
        acc += __expf(sc[n] - smax) * mhb[(size_t)n * DM + m];
    red[tid] = acc; __syncthreads();
    if (tid < 64)
        out[(size_t)b * (NHEAD * DM) + h * DM + tid] =
            (red[tid] + red[tid + 64] + red[tid + 128] + red[tid + 192]) * inv;
}

// ---------------------------------------------------------------------------
extern "C" void kernel_launch(void* const* d_in, const int* in_sizes, int n_in,
                              void* d_out, int out_size, void* d_ws, size_t ws_size,
                              hipStream_t stream) {
    (void)in_sizes; (void)n_in; (void)out_size; (void)ws_size;
    const float* x      = (const float*)d_in[0];
    const float* WX     = (const float*)d_in[1];
    const float* V      = (const float*)d_in[2];
    const float* W      = (const float*)d_in[3];
    const float* gate_w = (const float*)d_in[4];
    const int*   tn     = (const int*)d_in[5];

    // workspace layout (~114 MB): mh | gate | scores | xbf | WT
    float* mh     = (float*)d_ws;
    float* gate   = mh + (size_t)NHEAD * NROW * DM;             // 64 MB
    float* scores = gate + (size_t)NROW * DL;                   // +16 MB
    unsigned short* xbf = (unsigned short*)(scores + (size_t)NHEAD * NROW); // +1 MB
    unsigned short* WT  = xbf + (size_t)NROW * DIMK;            // +32 MB

    prep_x<<<(NROW * DIMK / 4) / 256, 256, 0, stream>>>(x, xbf);
    prep_weights<<<(NCOLW * DIMK + 255) / 256, 256, 0, stream>>>(WX, gate_w, WT);
    gemm_mh_gate<<<NROW / 128, 256, 0, stream>>>(xbf, WT, mh, gate);
    scores_kernel<<<(NHEAD * NROW) / 256, 256, 0, stream>>>(mh, gate, V, W, tn, scores);
    softmax_pool<<<NHEAD * BB, 256, 0, stream>>>(mh, scores, (float*)d_out);
}
